// LlamaSelfAttention_73899207295436
// MI455X (gfx1250) — matrix operand order repaired and checked by reading, not compiled, not run on hardware
//
#include <hip/hip_runtime.h>
#include <hip/hip_bf16.h>

typedef unsigned short u16;
typedef __bf16 v16bf __attribute__((ext_vector_type(16)));
typedef float  v8f   __attribute__((ext_vector_type(8)));

union Frag { uint4 u[2]; v16bf v; };

#define C_B   2
#define C_S   2048
#define C_D   2048
#define C_HQ  32
#define C_HKV 8
#define C_HD  64

// ---- CDNA5 async global->LDS path (guarded; falls back to sync staging) ----
#if defined(__gfx1250__) && __has_builtin(__builtin_amdgcn_global_load_async_to_lds_b128)
#define HAVE_ASYNC_LDS 1
typedef int v4i_vs __attribute__((vector_size(16)));
typedef __attribute__((address_space(1))) v4i_vs as1_v4i;
typedef __attribute__((address_space(3))) v4i_vs as3_v4i;
template <int OFF>
__device__ __forceinline__ void async_ld16(const u16* g, u16* l) {
    __builtin_amdgcn_global_load_async_to_lds_b128(
        (as1_v4i*)g, (as3_v4i*)l, OFF, 0);
}
#if __has_builtin(__builtin_amdgcn_s_wait_asynccnt)
#define WAIT_ASYNC() __builtin_amdgcn_s_wait_asynccnt(0)
#else
#define WAIT_ASYNC() asm volatile("s_wait_asynccnt 0x0" ::: "memory")
#endif
#else
#define HAVE_ASYNC_LDS 0
#define WAIT_ASYNC()
#endif

__device__ __forceinline__ u16 f2bf(float f) {
    unsigned int u = __float_as_uint(f);
    u += 0x7FFFu + ((u >> 16) & 1u);
    return (u16)(u >> 16);
}

__device__ __forceinline__ v8f wmma_bf16(const Frag& a, const Frag& b, v8f c) {
    return __builtin_amdgcn_wmma_f32_16x16x32_bf16(false, a.v, false, b.v,
                                                   (short)0, c, false, false);
}

// ---------------------------------------------------------------- fp32 -> bf16
__global__ __launch_bounds__(256) void cvt_bf16_kernel(const float* __restrict__ in,
                                                       u16* __restrict__ out, int n) {
    int i = blockIdx.x * 256 + threadIdx.x;
    if (i < n) out[i] = f2bf(in[i]);
}

// ------------------------------------------- fp32 [K,N] -> bf16 transposed [N,K]
__global__ __launch_bounds__(256)
void cvt_transpose_kernel(const float* __restrict__ in, u16* __restrict__ out,
                          int K, int N) {
    __shared__ __align__(16) u16 t[64][72];
    const int tid = threadIdx.x;
    const int k0 = blockIdx.x * 64, n0 = blockIdx.y * 64;
    const int r  = tid >> 2;
    const int c0 = (tid & 3) * 16;
    #pragma unroll
    for (int e = 0; e < 4; ++e) {
        float4 v = *(const float4*)&in[(size_t)(k0 + r) * N + n0 + c0 + e * 4];
        t[c0 + e * 4 + 0][r] = f2bf(v.x);
        t[c0 + e * 4 + 1][r] = f2bf(v.y);
        t[c0 + e * 4 + 2][r] = f2bf(v.z);
        t[c0 + e * 4 + 3][r] = f2bf(v.w);
    }
    __syncthreads();
    const int rn = tid >> 2;
    const int ck = (tid & 3) * 16;
    *(uint4*)&out[(size_t)(n0 + rn) * K + k0 + ck]     = *(uint4*)&t[rn][ck];
    *(uint4*)&out[(size_t)(n0 + rn) * K + k0 + ck + 8] = *(uint4*)&t[rn][ck + 8];
}

// ---------------------------------------------------------------- projection GEMM
// A[M,K] bf16, Wt[N,K] bf16 (pre-transposed). Block 256 thr; tile 128x128;
// wave tile 32x64. Double-buffered LDS (async global->LDS when available).
// Epilogue: optional RoPE; store bf16 [B,H,S,HD] (v_mode=0) or [B,H,HD,S] (v_mode=1).
__global__ __launch_bounds__(256)
void gemm_proj_kernel(const u16* __restrict__ A, const u16* __restrict__ Wt,
                      u16* __restrict__ Out, int N, int K, int nheads,
                      int do_rope, int v_mode) {
    __shared__ __align__(16) u16 lA [2][128][40];
    __shared__ __align__(16) u16 lBT[2][128][40];

    const int tid  = threadIdx.x;
    const int wave = tid >> 5;
    const int lane = tid & 31;
    const int half = lane >> 4;
    const int l16  = lane & 15;
    const int wm   = wave >> 1;
    const int wn   = wave & 1;
    const int m0   = blockIdx.x * 128;
    const int n0   = blockIdx.y * 128;

    v8f acc[2][4];
    #pragma unroll
    for (int mi = 0; mi < 2; ++mi)
        #pragma unroll
        for (int t = 0; t < 4; ++t) acc[mi][t] = (v8f){0,0,0,0,0,0,0,0};

    const int lr = tid >> 1;
    const int lc = (tid & 1) * 16;
    const int ntiles = K >> 5;

#if HAVE_ASYNC_LDS
    auto stage = [&](int buf, int k0) {
        const u16* ap = &A [(size_t)(m0 + lr) * K + k0 + lc];
        const u16* bp = &Wt[(size_t)(n0 + lr) * K + k0 + lc];
        async_ld16<0> (ap, &lA [buf][lr][lc]);
        async_ld16<16>(ap, &lA [buf][lr][lc]);
        async_ld16<0> (bp, &lBT[buf][lr][lc]);
        async_ld16<16>(bp, &lBT[buf][lr][lc]);
    };
    stage(0, 0);
    WAIT_ASYNC();
    __syncthreads();
#else
    uint4 pa0{}, pa1{}, pb0{}, pb1{};
    auto fetch = [&](int k0) {
        const u16* ap = &A [(size_t)(m0 + lr) * K + k0 + lc];
        const u16* bp = &Wt[(size_t)(n0 + lr) * K + k0 + lc];
        pa0 = *(const uint4*)ap; pa1 = *(const uint4*)(ap + 8);
        pb0 = *(const uint4*)bp; pb1 = *(const uint4*)(bp + 8);
    };
    fetch(0);
    *(uint4*)&lA [0][lr][lc] = pa0; *(uint4*)&lA [0][lr][lc + 8] = pa1;
    *(uint4*)&lBT[0][lr][lc] = pb0; *(uint4*)&lBT[0][lr][lc + 8] = pb1;
    __syncthreads();
#endif

    for (int i = 0; i < ntiles; ++i) {
        const int cur = i & 1;
#if HAVE_ASYNC_LDS
        if (i + 1 < ntiles) stage(cur ^ 1, (i + 1) << 5);
#else
        if (i + 1 < ntiles) fetch((i + 1) << 5);
#endif
        Frag a[2], bfr[4];
        #pragma unroll
        for (int mi = 0; mi < 2; ++mi) {
            int row = wm * 32 + mi * 16 + l16;
            a[mi].u[0] = *(const uint4*)&lA[cur][row][half * 8];
            a[mi].u[1] = *(const uint4*)&lA[cur][row][half * 8 + 16];
        }
        #pragma unroll
        for (int t = 0; t < 4; ++t) {
            int brow = wn * 64 + t * 16 + l16;
            bfr[t].u[0] = *(const uint4*)&lBT[cur][brow][half * 8];
            bfr[t].u[1] = *(const uint4*)&lBT[cur][brow][half * 8 + 16];
        }
        #pragma unroll
        for (int mi = 0; mi < 2; ++mi)
            #pragma unroll
            for (int t = 0; t < 4; ++t)
                acc[mi][t] = wmma_bf16(a[mi], bfr[t], acc[mi][t]);

#if HAVE_ASYNC_LDS
        if (i + 1 < ntiles) WAIT_ASYNC();
#else
        if (i + 1 < ntiles) {
            const int nxt = cur ^ 1;
            *(uint4*)&lA [nxt][lr][lc] = pa0; *(uint4*)&lA [nxt][lr][lc + 8] = pa1;
            *(uint4*)&lBT[nxt][lr][lc] = pb0; *(uint4*)&lBT[nxt][lr][lc + 8] = pb1;
        }
#endif
        __syncthreads();
    }

    const int hgl = (n0 + wn * 64) >> 6;
    #pragma unroll
    for (int mi = 0; mi < 2; ++mi) {
        #pragma unroll
        for (int t = 0; t < 4; ++t) {
            #pragma unroll
            for (int r = 0; r < 8; ++r) {
                int m = m0 + wm * 32 + mi * 16 + r + half * 8;
                int d = t * 16 + l16;
                float val = acc[mi][t][r];
                if (do_rope) {
                    float partner = acc[mi][t ^ 2][r];
                    float rot = (t < 2) ? -partner : partner;
                    int s = m & (C_S - 1);
                    float invf = __expf((float)(d & 31) * (-9.210340371976184f / 32.0f));
                    float sn, cs;
                    __sincosf((float)s * invf, &sn, &cs);
                    val = val * cs + rot * sn;
                }
                int b = m >> 11;
                int s = m & (C_S - 1);
                size_t off = v_mode
                    ? ((((size_t)b * nheads + hgl) * C_HD) + d) * C_S + s
                    : ((((size_t)b * nheads + hgl) * C_S) + s) * C_HD + d;
                Out[off] = f2bf(val);
            }
        }
    }
}

// ---------------------------------------------------------------- output GEMM
__global__ __launch_bounds__(256)
void gemm_out_kernel(const u16* __restrict__ A, const u16* __restrict__ Wt,
                     float* __restrict__ Cout, int N, int K) {
    __shared__ __align__(16) u16 lA [2][128][40];
    __shared__ __align__(16) u16 lBT[2][128][40];

    const int tid  = threadIdx.x;
    const int wave = tid >> 5;
    const int lane = tid & 31;
    const int half = lane >> 4;
    const int l16  = lane & 15;
    const int wm   = wave >> 1;
    const int wn   = wave & 1;
    const int m0   = blockIdx.x * 128;
    const int n0   = blockIdx.y * 128;

    v8f acc[2][4];
    #pragma unroll
    for (int mi = 0; mi < 2; ++mi)
        #pragma unroll
        for (int t = 0; t < 4; ++t) acc[mi][t] = (v8f){0,0,0,0,0,0,0,0};

    const int lr = tid >> 1;
    const int lc = (tid & 1) * 16;
    const int ntiles = K >> 5;

#if HAVE_ASYNC_LDS
    auto stage = [&](int buf, int k0) {
        const u16* ap = &A [(size_t)(m0 + lr) * K + k0 + lc];
        const u16* bp = &Wt[(size_t)(n0 + lr) * K + k0 + lc];
        async_ld16<0> (ap, &lA [buf][lr][lc]);
        async_ld16<16>(ap, &lA [buf][lr][lc]);
        async_ld16<0> (bp, &lBT[buf][lr][lc]);
        async_ld16<16>(bp, &lBT[buf][lr][lc]);
    };
    stage(0, 0);
    WAIT_ASYNC();
    __syncthreads();
#else
    uint4 pa0{}, pa1{}, pb0{}, pb1{};
    auto fetch = [&](int k0) {
        const u16* ap = &A [(size_t)(m0 + lr) * K + k0 + lc];
        const u16* bp = &Wt[(size_t)(n0 + lr) * K + k0 + lc];
        pa0 = *(const uint4*)ap; pa1 = *(const uint4*)(ap + 8);
        pb0 = *(const uint4*)bp; pb1 = *(const uint4*)(bp + 8);
    };
    fetch(0);
    *(uint4*)&lA [0][lr][lc] = pa0; *(uint4*)&lA [0][lr][lc + 8] = pa1;
    *(uint4*)&lBT[0][lr][lc] = pb0; *(uint4*)&lBT[0][lr][lc + 8] = pb1;
    __syncthreads();
#endif

    for (int i = 0; i < ntiles; ++i) {
        const int cur = i & 1;
#if HAVE_ASYNC_LDS
        if (i + 1 < ntiles) stage(cur ^ 1, (i + 1) << 5);
#else
        if (i + 1 < ntiles) fetch((i + 1) << 5);
#endif
        Frag a[2], bfr[4];
        #pragma unroll
        for (int mi = 0; mi < 2; ++mi) {
            int row = wm * 32 + mi * 16 + l16;
            a[mi].u[0] = *(const uint4*)&lA[cur][row][half * 8];
            a[mi].u[1] = *(const uint4*)&lA[cur][row][half * 8 + 16];
        }
        #pragma unroll
        for (int t = 0; t < 4; ++t) {
            int brow = wn * 64 + t * 16 + l16;
            bfr[t].u[0] = *(const uint4*)&lBT[cur][brow][half * 8];
            bfr[t].u[1] = *(const uint4*)&lBT[cur][brow][half * 8 + 16];
        }
        #pragma unroll
        for (int mi = 0; mi < 2; ++mi)
            #pragma unroll
            for (int t = 0; t < 4; ++t)
                acc[mi][t] = wmma_bf16(a[mi], bfr[t], acc[mi][t]);

#if HAVE_ASYNC_LDS
        if (i + 1 < ntiles) WAIT_ASYNC();
#else
        if (i + 1 < ntiles) {
            const int nxt = cur ^ 1;
            *(uint4*)&lA [nxt][lr][lc] = pa0; *(uint4*)&lA [nxt][lr][lc + 8] = pa1;
            *(uint4*)&lBT[nxt][lr][lc] = pb0; *(uint4*)&lBT[nxt][lr][lc + 8] = pb1;
        }
#endif
        __syncthreads();
    }

    #pragma unroll
    for (int mi = 0; mi < 2; ++mi)
        #pragma unroll
        for (int t = 0; t < 4; ++t)
            #pragma unroll
            for (int r = 0; r < 8; ++r) {
                int m = m0 + wm * 32 + mi * 16 + r + half * 8;
                int n = n0 + wn * 64 + t * 16 + l16;
                Cout[(size_t)m * N + n] = acc[mi][t][r];
            }
}

// ---------------------------------------------------------------- flash attention
// grid = (B*HQ, S/64), block = 128 (4 waves, 16 q rows each).
// Q,K bf16 [B,H,S,HD] (RoPE applied); V bf16 [B,HKV,HD,S] (pre-transposed).
__global__ __launch_bounds__(128)
void flash_attn_kernel(const u16* __restrict__ Q, const u16* __restrict__ Km,
                       const u16* __restrict__ Vt, u16* __restrict__ AO) {
#if HAVE_ASYNC_LDS
    __shared__ __align__(16) u16 lK [2][64][72];
    __shared__ __align__(16) u16 lVT[2][64][72];
#else
    __shared__ __align__(16) u16 lK [1][64][72];
    __shared__ __align__(16) u16 lVT[1][64][72];
#endif
    __shared__ __align__(16) u16 lP[4][16][72];

    const int tid  = threadIdx.x;
    const int wave = tid >> 5;
    const int lane = tid & 31;
    const int half = lane >> 4;
    const int l16  = lane & 15;

    const int bh  = blockIdx.x;
    const int b   = bh / C_HQ;
    const int h   = bh % C_HQ;
    const int hkv = h / (C_HQ / C_HKV);
    const int qt  = blockIdx.y;
    const int q0  = qt * 64;

    const u16* Qb  = Q  + ((size_t)b * C_HQ  + h)   * C_S  * C_HD;
    const u16* Kb  = Km + ((size_t)b * C_HKV + hkv) * C_S  * C_HD;
    const u16* Vtb = Vt + ((size_t)b * C_HKV + hkv) * C_HD * C_S;

    Frag qf[2];
    {
        int qrow = q0 + wave * 16 + l16;
        #pragma unroll
        for (int f = 0; f < 2; ++f) {
            int kb = f * 32 + half * 8;
            qf[f].u[0] = *(const uint4*)&Qb[(size_t)qrow * C_HD + kb];
            qf[f].u[1] = *(const uint4*)&Qb[(size_t)qrow * C_HD + kb + 16];
        }
    }

    v8f oacc[4];
    #pragma unroll
    for (int t = 0; t < 4; ++t) oacc[t] = (v8f){0,0,0,0,0,0,0,0};
    float mrow[8], lrow[8];
    #pragma unroll
    for (int r = 0; r < 8; ++r) { mrow[r] = -3.0e38f; lrow[r] = 0.0f; }

    const int lr = tid >> 1;          // 0..63
    const int lc = (tid & 1) * 32;    // 0 / 32

#if HAVE_ASYNC_LDS
    auto stage_kv = [&](int buf, int k0) {
        #pragma unroll
        for (int e = 0; e < 4; ++e) {
            async_ld16<0>(&Kb [(size_t)(k0 + lr) * C_HD + lc + e * 8],
                          &lK [buf][lr][lc + e * 8]);
            async_ld16<0>(&Vtb[(size_t)lr * C_S + k0 + lc + e * 8],
                          &lVT[buf][lr][lc + e * 8]);
        }
    };
    stage_kv(0, 0);
#else
    uint4 pk[4], pv[4];
    auto fetch_kv = [&](int k0) {
        #pragma unroll
        for (int e = 0; e < 4; ++e) {
            pk[e] = *(const uint4*)&Kb [(size_t)(k0 + lr) * C_HD + lc + e * 8];
            pv[e] = *(const uint4*)&Vtb[(size_t)lr * C_S + k0 + lc + e * 8];
        }
    };
    fetch_kv(0);
#endif

    for (int kt = 0; kt <= qt; ++kt) {
        const int k0 = kt * 64;
#if HAVE_ASYNC_LDS
        const int cur = kt & 1;
        WAIT_ASYNC();
        __syncthreads();
        if (kt < qt) stage_kv(cur ^ 1, k0 + 64);
#else
        const int cur = 0;
        #pragma unroll
        for (int e = 0; e < 4; ++e) {
            *(uint4*)&lK [0][lr][lc + e * 8] = pk[e];
            *(uint4*)&lVT[0][lr][lc + e * 8] = pv[e];
        }
        if (kt < qt) fetch_kv(k0 + 64);
        __syncthreads();
#endif

        // scores = Q K^T
        v8f sc[4];
        #pragma unroll
        for (int t = 0; t < 4; ++t) sc[t] = (v8f){0,0,0,0,0,0,0,0};
        #pragma unroll
        for (int f = 0; f < 2; ++f) {
            Frag bfr[4];
            #pragma unroll
            for (int t = 0; t < 4; ++t) {
                int j  = t * 16 + l16;
                int kb = f * 32 + half * 8;
                bfr[t].u[0] = *(const uint4*)&lK[cur][j][kb];
                bfr[t].u[1] = *(const uint4*)&lK[cur][j][kb + 16];
            }
            #pragma unroll
            for (int t = 0; t < 4; ++t) sc[t] = wmma_bf16(qf[f], bfr[t], sc[t]);
        }

        const float scale = 0.125f;
        #pragma unroll
        for (int t = 0; t < 4; ++t)
            #pragma unroll
            for (int r = 0; r < 8; ++r) {
                float v = sc[t][r] * scale;
                if (kt == qt) {
                    int kg = k0 + t * 16 + l16;
                    int qg = q0 + wave * 16 + r + half * 8;
                    if (kg > qg) v = -3.0e38f;
                }
                sc[t][r] = v;
            }

        #pragma unroll
        for (int r = 0; r < 8; ++r) {
            float mt = fmaxf(fmaxf(sc[0][r], sc[1][r]), fmaxf(sc[2][r], sc[3][r]));
            #pragma unroll
            for (int o = 1; o < 16; o <<= 1) mt = fmaxf(mt, __shfl_xor(mt, o, 32));
            float mnew = fmaxf(mrow[r], mt);
            float corr = __expf(mrow[r] - mnew);
            mrow[r] = mnew;
            float rs = 0.0f;
            #pragma unroll
            for (int t = 0; t < 4; ++t) {
                float p = __expf(sc[t][r] - mnew);
                sc[t][r] = p;
                rs += p;
            }
            #pragma unroll
            for (int o = 1; o < 16; o <<= 1) rs += __shfl_xor(rs, o, 32);
            lrow[r] = lrow[r] * corr + rs;
            #pragma unroll
            for (int t = 0; t < 4; ++t) oacc[t][r] *= corr;
        }

        #pragma unroll
        for (int t = 0; t < 4; ++t)
            #pragma unroll
            for (int r = 0; r < 8; ++r)
                lP[wave][r + half * 8][t * 16 + l16] = f2bf(sc[t][r]);
        __syncthreads();

        #pragma unroll
        for (int f = 0; f < 2; ++f) {
            Frag pa, vb[4];
            int kb = f * 32 + half * 8;
            pa.u[0] = *(const uint4*)&lP[wave][l16][kb];
            pa.u[1] = *(const uint4*)&lP[wave][l16][kb + 16];
            #pragma unroll
            for (int t = 0; t < 4; ++t) {
                int dcol = t * 16 + l16;
                int jb = f * 32 + half * 8;
                vb[t].u[0] = *(const uint4*)&lVT[cur][dcol][jb];
                vb[t].u[1] = *(const uint4*)&lVT[cur][dcol][jb + 16];
            }
            #pragma unroll
            for (int t = 0; t < 4; ++t) oacc[t] = wmma_bf16(pa, vb[t], oacc[t]);
        }
#if !HAVE_ASYNC_LDS
        __syncthreads();   // single-buffer reuse barrier
#endif
    }

    #pragma unroll
    for (int t = 0; t < 4; ++t)
        #pragma unroll
        for (int r = 0; r < 8; ++r) {
            int q = q0 + wave * 16 + r + half * 8;
            int d = t * 16 + l16;
            float val = oacc[t][r] / lrow[r];
            size_t off = (((size_t)b * C_S + q) * C_HQ + h) * C_HD + d;
            AO[off] = f2bf(val);
        }
}

// ---------------------------------------------------------------- launch
extern "C" void kernel_launch(void* const* d_in, const int* in_sizes, int n_in,
                              void* d_out, int out_size, void* d_ws, size_t ws_size,
                              hipStream_t stream) {
    (void)in_sizes; (void)n_in; (void)out_size; (void)ws_size;
    const float* x  = (const float*)d_in[0];
    const float* Wq = (const float*)d_in[1];
    const float* Wk = (const float*)d_in[2];
    const float* Wv = (const float*)d_in[3];
    const float* Wo = (const float*)d_in[4];
    float* out = (float*)d_out;

    char* ws = (char*)d_ws;
    const size_t NX = (size_t)C_B * C_S * C_D;           // 8388608
    u16* xb   = (u16*)(ws);
    u16* Wqbt = (u16*)(ws + 16777216);
    u16* Wkbt = (u16*)(ws + 25165824);
    u16* Wvbt = (u16*)(ws + 27262976);
    u16* Wobt = (u16*)(ws + 29360128);
    u16* Qb   = (u16*)(ws + 37748736);
    u16* Kb   = (u16*)(ws + 54525952);
    u16* Vbt  = (u16*)(ws + 58720256);
    u16* AOb  = (u16*)(ws + 62914560);

    cvt_bf16_kernel<<<dim3((unsigned)(NX / 256)), 256, 0, stream>>>(x, xb, (int)NX);
    cvt_transpose_kernel<<<dim3(32, 32), 256, 0, stream>>>(Wq, Wqbt, C_D, C_HQ * C_HD);
    cvt_transpose_kernel<<<dim3(32, 8),  256, 0, stream>>>(Wk, Wkbt, C_D, C_HKV * C_HD);
    cvt_transpose_kernel<<<dim3(32, 8),  256, 0, stream>>>(Wv, Wvbt, C_D, C_HKV * C_HD);
    cvt_transpose_kernel<<<dim3(32, 32), 256, 0, stream>>>(Wo, Wobt, C_D, C_D);

    const int M = C_B * C_S;  // 4096
    gemm_proj_kernel<<<dim3(M / 128, (C_HQ * C_HD) / 128), 256, 0, stream>>>(
        xb, Wqbt, Qb, C_HQ * C_HD, C_D, C_HQ, 1, 0);
    gemm_proj_kernel<<<dim3(M / 128, (C_HKV * C_HD) / 128), 256, 0, stream>>>(
        xb, Wkbt, Kb, C_HKV * C_HD, C_D, C_HKV, 1, 0);
    gemm_proj_kernel<<<dim3(M / 128, (C_HKV * C_HD) / 128), 256, 0, stream>>>(
        xb, Wvbt, Vbt, C_HKV * C_HD, C_D, C_HKV, 0, 1);

    flash_attn_kernel<<<dim3(C_B * C_HQ, C_S / 64), 128, 0, stream>>>(Qb, Kb, Vbt, AOb);

    gemm_out_kernel<<<dim3(M / 128, C_D / 128), 256, 0, stream>>>(AOb, Wobt, out, C_D, C_D);
}
